// GraphEncoder_20555713479231
// MI455X (gfx1250) — matrix operand order repaired and checked by reading, not compiled, not run on hardware
//
#include <hip/hip_runtime.h>
#include <math.h>

typedef __attribute__((ext_vector_type(16))) _Float16 v16h;
typedef __attribute__((ext_vector_type(8)))  float    v8f;

#define HID 128  // H*D == 128 for every layer

// ---------------------------------------------------------------- utilities
__device__ __forceinline__ float atomicMaxF(float* addr, float val) {
  // int-bit ordering trick: works across mixed signs, init must be -inf
  if (val >= 0.0f)
    return __int_as_float(atomicMax((int*)addr, __float_as_int(val)));
  else
    return __uint_as_float(atomicMin((unsigned int*)addr, __float_as_uint(val)));
}

__global__ void ge_fill_kernel(float* __restrict__ p, float v, long n) {
  long i = (long)blockIdx.x * blockDim.x + threadIdx.x;
  if (i < n) p[i] = v;
}

__global__ void ge_embed_kernel(const int* __restrict__ nt,
                                const float* __restrict__ emb,
                                float* __restrict__ x, int N) {
  int i = blockIdx.x * blockDim.x + threadIdx.x;
  if (i >= N * 16) return;
  int node = i >> 4, j = i & 15;
  x[i] = emb[nt[node] * 16 + j];
}

// ---------------------------------------------------------------- dual GEMM
// xl = x @ Wl, xr = x @ Wr.  x: [N][K] f32 row-major, Wl/Wr: [K][128] f32.
// One block = 16 rows; 8 waves, wave w owns output cols [16w, 16w+16).
// K is compile-time; the (rare) partial-row case is a block-uniform branch.
template <int K>
__global__ __launch_bounds__(256)
void ge_dual_gemm_wmma(const float* __restrict__ x, const float* __restrict__ Wl,
                       const float* __restrict__ Wr, float* __restrict__ xl,
                       float* __restrict__ xr, int N) {
  constexpr int NK = (K + 31) / 32;   // k-chunks of 32
  constexpr int KP = NK * 32;         // zero-padded K
  constexpr int PITCH = KP + 4;       // conflict-free column gathers
  __shared__ float xt[16 * PITCH];

  const int tid = threadIdx.x;
  const int m0 = blockIdx.x * 16;
  const bool full = (m0 + 16 <= N);   // uniform per block

  // stage A tile (unguarded fast path), zero-pad columns K..KP separately
  if (full) {
    for (int idx = tid; idx < 16 * K; idx += 256) {
      int r = idx / K, c = idx - r * K;
      xt[r * PITCH + c] = x[(long)(m0 + r) * K + c];
    }
  } else {
    for (int idx = tid; idx < 16 * K; idx += 256) {
      int r = idx / K, c = idx - r * K;
      int row = m0 + r;
      xt[r * PITCH + c] = (row < N) ? x[(long)row * K + c] : 0.0f;
    }
  }
  if constexpr (KP > K) {
    for (int idx = tid; idx < 16 * (KP - K); idx += 256) {
      int r = idx / (KP - K), c = K + idx - r * (KP - K);
      xt[r * PITCH + c] = 0.0f;
    }
  }
  __syncthreads();

  const int lane = tid & 31;
  const int wave = tid >> 5;
  const int n0 = wave * 16;
  const int lm = lane & 15;   // M index (A) / N index (B, C)
  const int hi = lane >> 4;   // upper-half-lane flag

  const float* WlB = Wl + n0 + lm;  // column base pointers
  const float* WrB = Wr + n0 + lm;

  v8f accL = {};
  v8f accR = {};

#pragma unroll
  for (int kc = 0; kc < NK; ++kc) {
    const int kb = kc * 32;

    // A fragment: 16x32 f16, k = 16*hi + s (matches B); pairs (k,k+1) -> ds_load_b64
    v16h a;
#pragma unroll
    for (int v = 0; v < 8; ++v) {
      int k = kb + 2 * v + hi * 16;
      float2 f = *(const float2*)&xt[lm * PITCH + k];
      a[2 * v]     = (_Float16)f.x;
      a[2 * v + 1] = (_Float16)f.y;
    }

    // B fragments: 32x16 f16 (lane<16: K 0-15; lane>=16: K 16-31), N = lm.
    // Branch-free: clamp address + 0/1 mask (folds away when K==128).
    v16h bl, br;
#pragma unroll
    for (int v = 0; v < 8; ++v) {
      int k = kb + hi * 16 + 2 * v;
      int  k0 = (k     < K) ? k     : 0;
      int  k1 = (k + 1 < K) ? k + 1 : 0;
      float q0 = (k     < K) ? 1.0f : 0.0f;
      float q1 = (k + 1 < K) ? 1.0f : 0.0f;
      bl[2 * v]     = (_Float16)(WlB[(long)k0 * HID] * q0);
      bl[2 * v + 1] = (_Float16)(WlB[(long)k1 * HID] * q1);
      br[2 * v]     = (_Float16)(WrB[(long)k0 * HID] * q0);
      br[2 * v + 1] = (_Float16)(WrB[(long)k1 * HID] * q1);
    }

    accL = __builtin_amdgcn_wmma_f32_16x16x32_f16(false, a, false, bl,
                                                  (short)0, accL, false, false);
    accR = __builtin_amdgcn_wmma_f32_16x16x32_f16(false, a, false, br,
                                                  (short)0, accR, false, false);
  }

  // C/D layout: row = m0 + v + 8*hi, col = n0 + lm.
  // Fast path: single base pointer, immediate offsets v*HID*4 (fits IOFFSET).
  float* outL = xl + (long)(m0 + 8 * hi) * HID + n0 + lm;
  float* outR = xr + (long)(m0 + 8 * hi) * HID + n0 + lm;
  if (full) {
#pragma unroll
    for (int v = 0; v < 8; ++v) {
      outL[v * HID] = accL[v];
      outR[v * HID] = accR[v];
    }
  } else {
#pragma unroll
    for (int v = 0; v < 8; ++v) {
      int row = m0 + v + 8 * hi;
      if (row < N) {
        outL[v * HID] = accL[v];
        outR[v * HID] = accR[v];
      }
    }
  }
}

// ---------------------------------------------------------------- edge phase
template <int H>
__device__ __forceinline__ void ge_edge_decode(long t, int E, int N,
                                               int& e, int& h, int& s, int& d,
                                               const int* __restrict__ ei) {
  e = (int)(t / H);              // H is 4 or 1 -> shift
  h = (int)(t - (long)e * H);
  if (e < E) { s = ei[e]; d = ei[E + e]; }
  else       { s = e - E; d = s; }
}

template <int H, int D>
__global__ void ge_edge_logit_kernel(const float* __restrict__ xl,
                                     const float* __restrict__ xr,
                                     const int* __restrict__ ei,
                                     const float* __restrict__ att,
                                     float* __restrict__ logit,
                                     float* __restrict__ lmax,
                                     int E, int N) {
  long t = (long)blockIdx.x * blockDim.x + threadIdx.x;
  long total = (long)(E + N) * H;
  if (t >= total) return;
  int e, h, s, d;
  ge_edge_decode<H>(t, E, N, e, h, s, d, ei);

  const float4* pl = (const float4*)(xl + (long)s * H * D + h * D);
  const float4* pr = (const float4*)(xr + (long)d * H * D + h * D);
  const float4* pa = (const float4*)(att + (long)h * D);
  float acc = 0.0f;
#pragma unroll
  for (int i = 0; i < D / 4; ++i) {
    float4 vl = pl[i], vr = pr[i], va = pa[i];
    float v0 = vl.x + vr.x; v0 = v0 > 0.0f ? v0 : 0.2f * v0;
    float v1 = vl.y + vr.y; v1 = v1 > 0.0f ? v1 : 0.2f * v1;
    float v2 = vl.z + vr.z; v2 = v2 > 0.0f ? v2 : 0.2f * v2;
    float v3 = vl.w + vr.w; v3 = v3 > 0.0f ? v3 : 0.2f * v3;
    acc += va.x * v0 + va.y * v1 + va.z * v2 + va.w * v3;
  }
  logit[t] = acc;
  atomicMaxF(&lmax[(long)d * H + h], acc);
}

template <int H>
__global__ void ge_edge_softmax_kernel(float* __restrict__ logit,  // in: logit, out: p
                                       const float* __restrict__ lmax,
                                       const int* __restrict__ ei,
                                       float* __restrict__ z,
                                       int E, int N) {
  long t = (long)blockIdx.x * blockDim.x + threadIdx.x;
  long total = (long)(E + N) * H;
  if (t >= total) return;
  int e, h, s, d;
  ge_edge_decode<H>(t, E, N, e, h, s, d, ei);
  float pv = __expf(logit[t] - lmax[(long)d * H + h]);
  logit[t] = pv;  // in-place: each thread owns index t
  atomicAdd(&z[(long)d * H + h], pv);
}

template <int H, int D>
__global__ void ge_edge_aggregate_kernel(const float* __restrict__ xl,
                                         const float* __restrict__ p,
                                         const float* __restrict__ z,
                                         const int* __restrict__ ei,
                                         float* __restrict__ agg,
                                         int E, int N) {
  long t = (long)blockIdx.x * blockDim.x + threadIdx.x;
  long total = (long)(E + N) * H;
  if (t >= total) return;
  int e, h, s, d;
  ge_edge_decode<H>(t, E, N, e, h, s, d, ei);
  float alpha = p[t] / z[(long)d * H + h];
  const float4* src = (const float4*)(xl + (long)s * H * D + h * D);
  float* dst = agg + (long)d * H * D + h * D;
  __builtin_prefetch(src, 0, 1);  // global_prefetch_b8
#pragma unroll
  for (int i = 0; i < D / 4; ++i) {
    float4 v = src[i];
    atomicAdd(&dst[4 * i + 0], alpha * v.x);
    atomicAdd(&dst[4 * i + 1], alpha * v.y);
    atomicAdd(&dst[4 * i + 2], alpha * v.z);
    atomicAdd(&dst[4 * i + 3], alpha * v.w);
  }
}

// ------------------------- post: bias + concat/mean + residual + LN + ReLU
__global__ __launch_bounds__(256)
void ge_post_kernel(const float* __restrict__ agg, const float* __restrict__ xprev,
                    const float* __restrict__ bias, const float* __restrict__ g,
                    const float* __restrict__ beta, float* __restrict__ xout,
                    int N, int H, int D, int concat, int hasRes) {
  int lane = threadIdx.x & 31;
  int wv = threadIdx.x >> 5;
  int node = blockIdx.x * 8 + wv;
  if (node >= N) return;

  float val[4];
#pragma unroll
  for (int j = 0; j < 4; ++j) {
    int col = lane * 4 + j;  // 128 output cols for every layer
    float v;
    if (concat) {
      v = agg[(long)node * H * D + col];
    } else {
      v = 0.0f;
      for (int hh = 0; hh < H; ++hh) v += agg[(long)node * H * D + hh * D + col];
      v *= (1.0f / (float)H);
    }
    v += bias[col];
    if (hasRes) v += xprev[(long)node * HID + col];
    val[j] = v;
  }

  float s = val[0] + val[1] + val[2] + val[3];
  for (int o = 16; o > 0; o >>= 1) s += __shfl_xor(s, o, 32);
  float mu = s * (1.0f / 128.0f);

  float vs = 0.0f;
#pragma unroll
  for (int j = 0; j < 4; ++j) { float dd = val[j] - mu; vs += dd * dd; }
  for (int o = 16; o > 0; o >>= 1) vs += __shfl_xor(vs, o, 32);
  float inv = rsqrtf(vs * (1.0f / 128.0f) + 1e-5f);

  float4 outv;
  float* ov = (float*)&outv;
#pragma unroll
  for (int j = 0; j < 4; ++j) {
    int col = lane * 4 + j;
    float y = (val[j] - mu) * inv * g[col] + beta[col];
    ov[j] = y > 0.0f ? y : 0.0f;
  }
  *(float4*)(xout + (long)node * HID + lane * 4) = outv;
}

// ---------------------------------------------------------------- launch
extern "C" void kernel_launch(void* const* d_in, const int* in_sizes, int n_in,
                              void* d_out, int out_size, void* d_ws, size_t ws_size,
                              hipStream_t stream) {
  const int*   node_types = (const int*)d_in[0];
  const int*   edge_index = (const int*)d_in[1];
  const float* emb        = (const float*)d_in[2];
  const float* ln_g       = (const float*)d_in[19];
  const float* ln_b       = (const float*)d_in[20];

  const int N = in_sizes[0];
  const int E = in_sizes[1] / 2;

  const long nHD = (long)N * HID;
  const long eH  = (long)(E + N) * 4;  // max H == 4

  float* ws    = (float*)d_ws;
  float* xbuf0 = ws;                 // current features (layer 0: N x 16)
  float* xbuf1 = xbuf0 + nHD;        // next features
  float* xl    = xbuf1 + nHD;        // source transform
  float* xr    = xl + nHD;           // target transform; aliased as agg later
  float* agg   = xr;                 // safe: xr last read in edge_logit
  float* logit = xr + nHD;           // logit, then p in place
  float* lmax  = logit + eH;
  float* zbuf  = lmax + (long)N * 4;

  const int HV[4] = {4, 4, 4, 1};
  const int DV[4] = {32, 32, 32, 128};
  const int CC[4] = {1, 1, 1, 0};
  const int RS[4] = {0, 1, 1, 1};

  ge_embed_kernel<<<(N * 16 + 255) / 256, 256, 0, stream>>>(node_types, emb, xbuf0, N);

  float* xcur = xbuf0;
  float* xnext = xbuf1;
  for (int li = 0; li < 4; ++li) {
    const float* Wl   = (const float*)d_in[3 + 4 * li];
    const float* Wr   = (const float*)d_in[4 + 4 * li];
    const float* att  = (const float*)d_in[5 + 4 * li];
    const float* bias = (const float*)d_in[6 + 4 * li];
    const int H = HV[li], D = DV[li];

    if (li == 0)
      ge_dual_gemm_wmma<16><<<(N + 15) / 16, 256, 0, stream>>>(xcur, Wl, Wr, xl, xr, N);
    else
      ge_dual_gemm_wmma<128><<<(N + 15) / 16, 256, 0, stream>>>(xcur, Wl, Wr, xl, xr, N);

    ge_fill_kernel<<<(int)(((long)N * H + 255) / 256), 256, 0, stream>>>(
        lmax, -INFINITY, (long)N * H);

    const long totEH = (long)(E + N) * H;
    const int gb = (int)((totEH + 255) / 256);

    if (li < 3) {
      ge_edge_logit_kernel<4, 32><<<gb, 256, 0, stream>>>(xl, xr, edge_index, att,
                                                          logit, lmax, E, N);
    } else {
      ge_edge_logit_kernel<1, 128><<<gb, 256, 0, stream>>>(xl, xr, edge_index, att,
                                                           logit, lmax, E, N);
    }

    // after edge_logit, xr is dead -> reuse as agg
    ge_fill_kernel<<<(int)((nHD + 255) / 256), 256, 0, stream>>>(agg, 0.0f, nHD);
    ge_fill_kernel<<<(int)(((long)N * H + 255) / 256), 256, 0, stream>>>(
        zbuf, 0.0f, (long)N * H);

    if (li < 3) {
      ge_edge_softmax_kernel<4><<<gb, 256, 0, stream>>>(logit, lmax, edge_index,
                                                        zbuf, E, N);
      ge_edge_aggregate_kernel<4, 32><<<gb, 256, 0, stream>>>(xl, logit, zbuf,
                                                              edge_index, agg, E, N);
    } else {
      ge_edge_softmax_kernel<1><<<gb, 256, 0, stream>>>(logit, lmax, edge_index,
                                                        zbuf, E, N);
      ge_edge_aggregate_kernel<1, 128><<<gb, 256, 0, stream>>>(xl, logit, zbuf,
                                                               edge_index, agg, E, N);
    }

    float* outp = (li == 3) ? (float*)d_out : xnext;
    ge_post_kernel<<<(N + 7) / 8, 256, 0, stream>>>(agg, xcur, bias,
                                                    ln_g + (long)li * HID,
                                                    ln_b + (long)li * HID,
                                                    outp, N, H, D, CC[li], RS[li]);
    float* tmp = xcur; xcur = xnext; xnext = tmp;
  }
}
